// InfoNCELoss_22746146799616
// MI455X (gfx1250) — matrix operand order repaired and checked by reading, not compiled, not run on hardware
//
#include <hip/hip_runtime.h>
#include <hip/hip_bf16.h>

// ---------------------------------------------------------------------------
// InfoNCE (CPC) loss for MI455X / gfx1250.
//   B=8, T=256, D=256, 12 steps, 100 negatives, temp=0.1.
// Dense projection GEMMs run on v_wmma_f32_16x16x32_bf16 (fp32 accumulate),
// 16x64 register blocking per wave (4 independent accumulator chains) with
// double-buffered fragment loads. amdgpu_waves_per_eu(1) lifts the register
// pressure target so the scheduler can keep both buffers live and hoist
// next-chunk loads above current-chunk WMMAs.
// Working set (~30 MB) is L2-resident (192 MB): direct global->VGPR loads,
// no LDS/TDM staging needed.
// ---------------------------------------------------------------------------

#define BB    8
#define TT    256
#define DD    256
#define STEPS 12
#define NNEG  100
#define INV_TEMP 10.0f
#define EPSN  1e-12f

typedef __attribute__((ext_vector_type(8)))  __bf16 v8bf;
typedef __attribute__((ext_vector_type(16))) __bf16 v16bf;
typedef __attribute__((ext_vector_type(8)))  float  v8f;

__device__ __forceinline__ unsigned short f2bf(float f) {
  unsigned u = __float_as_uint(f);
  unsigned r = u + 0x7FFFu + ((u >> 16) & 1u);   // round-to-nearest-even
  return (unsigned short)(r >> 16);
}

// ---------------------------------------------------------------------------
// 1) fp32 -> bf16 for context and W; zero the scalar loss output.
// ---------------------------------------------------------------------------
__global__ void k_prep(const float* __restrict__ ctx, const float* __restrict__ W,
                       unsigned short* __restrict__ ctx_bf,
                       unsigned short* __restrict__ w_bf,
                       float* __restrict__ out) {
  const int nctx = BB * TT * DD;
  const int nw   = STEPS * DD * DD;
  int gid = blockIdx.x * blockDim.x + threadIdx.x;
  if (gid == 0) out[0] = 0.0f;
  for (int i = gid; i < nctx; i += gridDim.x * blockDim.x) ctx_bf[i] = f2bf(ctx[i]);
  for (int i = gid; i < nw;   i += gridDim.x * blockDim.x) w_bf[i]   = f2bf(W[i]);
}

// ---------------------------------------------------------------------------
// 2) L2-normalize every target row (one 256-thread block per row).
// ---------------------------------------------------------------------------
__device__ __forceinline__ float block_sum(float v, float* sbuf) {
  int tid = threadIdx.x;
  sbuf[tid] = v;
  __syncthreads();
  for (int s = blockDim.x >> 1; s > 0; s >>= 1) {
    if (tid < s) sbuf[tid] += sbuf[tid + s];
    __syncthreads();
  }
  float r = sbuf[0];
  __syncthreads();
  return r;
}

__global__ void k_tnorm(const float* __restrict__ tgt, float* __restrict__ posn) {
  __shared__ float sbuf[256];
  const int row = blockIdx.x;            // 0 .. B*T-1
  const int tid = threadIdx.x;           // 0 .. 255 == d
  float v = tgt[(size_t)row * DD + tid];
  float ss = block_sum(v * v, sbuf);
  float n = fmaxf(sqrtf(ss), EPSN);
  posn[(size_t)row * DD + tid] = v / n;
}

// ---------------------------------------------------------------------------
// 3) WMMA GEMM:  q[s, r, e] = sum_d ctx[r,d] * W[s,e,d] + b[s,e]
//    16x64 per wave, K=256 in 8 chunks of 32 -> 32 WMMA / wave.
//    Double-buffered fragments: loads for chunk k+32 are issued before the
//    WMMAs consuming chunk k. waves_per_eu(1) allows full VGPR budget.
//    Fragment layouts used here (contiguous k, same order in A and B):
//      A: lane L (m=L&15, h=L>>4) holds K in {16h..16h+15}
//      B: lane L (n=L&15, h=L>>4) holds K in {16h..16h+15}  (B[k][n]=W[n][k])
//      C/D: lane L holds (M = r + 8h, N = L&15) in VGPR r.
// ---------------------------------------------------------------------------
__device__ __forceinline__ void load_frags(const unsigned short* arow,
                                           const unsigned short* brow0,
                                           const unsigned short* brow1,
                                           const unsigned short* brow2,
                                           const unsigned short* brow3,
                                           int k0, int half,
                                           v16bf& a, v16bf& b0, v16bf& b1,
                                           v16bf& b2, v16bf& b3) {
  v8bf alo = *(const v8bf*)(arow + k0 + 16 * half);
  v8bf ahi = *(const v8bf*)(arow + k0 + 8 + 16 * half);
  a = __builtin_shufflevector(alo, ahi, 0, 1, 2, 3, 4, 5, 6, 7,
                              8, 9, 10, 11, 12, 13, 14, 15);
  b0 = *(const v16bf*)(brow0 + k0);
  b1 = *(const v16bf*)(brow1 + k0);
  b2 = *(const v16bf*)(brow2 + k0);
  b3 = *(const v16bf*)(brow3 + k0);
}

__global__ void
__launch_bounds__(32)
__attribute__((amdgpu_waves_per_eu(1)))
k_qgemm(const unsigned short* __restrict__ ctx_bf,  // [B*T, D]
        const unsigned short* __restrict__ w_bf,    // [S, D(e), D(d)]
        const float* __restrict__ bias,             // [S, D]
        float* __restrict__ q) {                    // [S, B*T, D]
  const int lane = threadIdx.x;        // 0..31
  const int mt   = blockIdx.x;         // 0..127  (16-row tiles of B*T)
  const int nt   = blockIdx.y;         // 0..3    (64-col blocks of D)
  const int s    = blockIdx.z;         // 0..11
  const int half = lane >> 4;
  const int l16  = lane & 15;

  const unsigned short* arow  = ctx_bf + (size_t)(mt * 16 + l16) * DD;
  const unsigned short* wb    = w_bf + (size_t)s * DD * DD;
  const unsigned short* brow0 = wb + (size_t)(nt * 64 +  0 + l16) * DD + 16 * half;
  const unsigned short* brow1 = wb + (size_t)(nt * 64 + 16 + l16) * DD + 16 * half;
  const unsigned short* brow2 = wb + (size_t)(nt * 64 + 32 + l16) * DD + 16 * half;
  const unsigned short* brow3 = wb + (size_t)(nt * 64 + 48 + l16) * DD + 16 * half;

  v8f c0 = {}, c1 = {}, c2 = {}, c3 = {};
  v16bf a, b0, b1, b2, b3;           // current fragments
  v16bf an, b0n, b1n, b2n, b3n;      // next fragments (double buffer)

  load_frags(arow, brow0, brow1, brow2, brow3, 0, half, a, b0, b1, b2, b3);

  #pragma unroll
  for (int k0 = 0; k0 < DD; k0 += 32) {
    if (k0 + 32 < DD) {
      load_frags(arow, brow0, brow1, brow2, brow3, k0 + 32, half,
                 an, b0n, b1n, b2n, b3n);
    }
    c0 = __builtin_amdgcn_wmma_f32_16x16x32_bf16(false, a, false, b0, (short)0, c0, false, false);
    c1 = __builtin_amdgcn_wmma_f32_16x16x32_bf16(false, a, false, b1, (short)0, c1, false, false);
    c2 = __builtin_amdgcn_wmma_f32_16x16x32_bf16(false, a, false, b2, (short)0, c2, false, false);
    c3 = __builtin_amdgcn_wmma_f32_16x16x32_bf16(false, a, false, b3, (short)0, c3, false, false);
    a = an; b0 = b0n; b1 = b1n; b2 = b2n; b3 = b3n;   // rotate (renamed away)
  }

  float* qbase = q + (size_t)s * (BB * TT) * DD;
  const float* bs = bias + s * DD + nt * 64 + l16;
  #pragma unroll
  for (int sub = 0; sub < 4; ++sub) {
    const float be = bs[sub * 16];
    const v8f& c = (sub == 0) ? c0 : (sub == 1) ? c1 : (sub == 2) ? c2 : c3;
    #pragma unroll
    for (int r = 0; r < 8; ++r) {
      int m = mt * 16 + r + 8 * half;
      qbase[(size_t)m * DD + nt * 64 + sub * 16 + l16] = c[r] + be;
    }
  }
}

// ---------------------------------------------------------------------------
// 4) Normalize q rows in place + fused positive similarity.
//    grid (B*T, STEPS), block 256.
// ---------------------------------------------------------------------------
__global__ void k_qnorm(float* __restrict__ q, const float* __restrict__ posn,
                        float* __restrict__ psim) {
  __shared__ float sbuf[256];
  const int row  = blockIdx.x;         // b*T + t
  const int s    = blockIdx.y;         // 0..11
  const int step = s + 1;
  const int t    = row % TT;
  if (t >= TT - step) return;          // outside valid T2 range

  const int tid = threadIdx.x;
  float* qrow = q + ((size_t)s * (BB * TT) + row) * DD;
  float v = qrow[tid];
  float ss = block_sum(v * v, sbuf);
  float n = fmaxf(sqrtf(ss), EPSN);
  float qn = v / n;
  qrow[tid] = qn;                       // normalized in place

  // positive: same b, time t+step -> flat row (row + step)
  float pd = qn * posn[(size_t)(row + step) * DD + tid];
  float dot = block_sum(pd, sbuf);
  if (tid == 0) psim[s * (BB * TT) + row] = dot * INV_TEMP;
}

// ---------------------------------------------------------------------------
// 5) Negatives + log-softmax + accumulate loss.
//    grid (B*T, STEPS), block 128. Deterministic hash replaces the threefry
//    stream (same distribution role; deterministic per call).
// ---------------------------------------------------------------------------
__device__ __forceinline__ unsigned mix32(unsigned h) {
  h ^= h >> 16; h *= 0x7FEB352Du;
  h ^= h >> 15; h *= 0x846CA68Bu;
  h ^= h >> 16;
  return h;
}

__global__ void k_negloss(const float* __restrict__ q,     // normalized
                          const float* __restrict__ posn,
                          const float* __restrict__ psim,
                          float* __restrict__ out) {
  __shared__ float qn_s[DD];
  __shared__ float logits[NNEG + 1];

  const int row  = blockIdx.x;
  const int s    = blockIdx.y;
  const int step = s + 1;
  const int T2   = TT - step;
  const int t    = row % TT;
  if (t >= T2) return;

  const int tid = threadIdx.x;          // 0..127
  const float* qrow = q + ((size_t)s * (BB * TT) + row) * DD;
  for (int d = tid; d < DD; d += blockDim.x) qn_s[d] = qrow[d];
  if (tid == 0) logits[0] = psim[s * (BB * TT) + row];
  __syncthreads();

  if (tid < NNEG) {
    unsigned h = mix32(0x9E3779B9u * (unsigned)step ^
                       0x85EBCA6Bu * (unsigned)row ^
                       0xC2B2AE35u * (unsigned)tid);
    int idx = (int)(h % (unsigned)(BB * T2));     // flat index into (B, T2)
    int bn  = idx / T2;
    int tn  = idx % T2;
    const float* nr = posn + (size_t)(bn * TT + tn + step) * DD;
    float dot = 0.0f;
    #pragma unroll 8
    for (int d = 0; d < DD; ++d) dot = fmaf(qn_s[d], nr[d], dot);
    logits[1 + tid] = dot * INV_TEMP;
  }
  __syncthreads();

  if (tid == 0) {
    float m = logits[0];
    for (int i = 1; i <= NNEG; ++i) m = fmaxf(m, logits[i]);
    float se = 0.0f;
    for (int i = 0; i <= NNEG; ++i) se += __expf(logits[i] - m);
    float lse = m + __logf(se);
    // per-(b,t,step) weight: mean over B*T2, then mean over 12 steps
    float w = 1.0f / ((float)(BB * T2) * (float)STEPS);
    atomicAdd(out, (lse - logits[0]) * w);
  }
}

// ---------------------------------------------------------------------------
// launch
// ---------------------------------------------------------------------------
extern "C" void kernel_launch(void* const* d_in, const int* in_sizes, int n_in,
                              void* d_out, int out_size, void* d_ws, size_t ws_size,
                              hipStream_t stream) {
  (void)in_sizes; (void)n_in; (void)out_size; (void)ws_size;

  const float* ctx  = (const float*)d_in[0];   // (B,T,D)
  const float* tgt  = (const float*)d_in[1];   // (B,T,D)
  const float* W    = (const float*)d_in[2];   // (S,D,D)
  const float* bias = (const float*)d_in[3];   // (S,D)
  float* out = (float*)d_out;

  // workspace layout (256B aligned)
  char* ws = (char*)d_ws;
  unsigned short* ctx_bf = (unsigned short*)(ws);            //  1.00 MB
  unsigned short* w_bf   = (unsigned short*)(ws + 1048576);  //  1.50 MB
  float* posn = (float*)(ws + 2621440);                      //  2.00 MB
  float* q    = (float*)(ws + 4718592);                      // 24.00 MB
  float* psim = (float*)(ws + 29884416);                     //  0.09 MB

  k_prep<<<512, 256, 0, stream>>>(ctx, W, ctx_bf, w_bf, out);
  k_tnorm<<<BB * TT, 256, 0, stream>>>(tgt, posn);

  dim3 gg(128, 4, STEPS);    // 16-row tiles x 64-col blocks x steps
  k_qgemm<<<gg, 32, 0, stream>>>(ctx_bf, w_bf, bias, q);

  dim3 gn(BB * TT, STEPS);
  k_qnorm<<<gn, 256, 0, stream>>>(q, posn, psim);
  k_negloss<<<gn, 128, 0, stream>>>(q, posn, psim, out);
}
